// MultiHeadAttention_69320772158274
// MI455X (gfx1250) — matrix operand order repaired and checked by reading, not compiled, not run on hardware
//
#include <hip/hip_runtime.h>
#include <math.h>

#define BATCH   4
#define SEQ     4096
#define HEADS   16
#define KSZ     128
#define DMODEL  2048
#define EPS     1e-6f
#define MASK_NEG (-1e9f)

// All four GEMMs are the same shape: hardcode so stores use immediate offsets.
#define GM 16384
#define GN 2048
#define GK 2048

#define TILE_M 128
#define TILE_N 128
#define TILE_K 32

typedef __attribute__((ext_vector_type(16))) __bf16 v16bf;
typedef __attribute__((ext_vector_type(8)))  float  v8f;

// fp32 -> bf16 round-to-nearest-even (bit trick)
__device__ __forceinline__ unsigned int f32_bf16_bits(float f) {
    unsigned int u = __float_as_uint(f);
    u += 0x7FFFu + ((u >> 16) & 1u);
    return u >> 16;
}
__device__ __forceinline__ unsigned int pack_bf16(float a, float b) {
    return f32_bf16_bits(a) | (f32_bf16_bits(b) << 16);
}

// LDS tiles hold packed bf16 pairs along K: row stride = TILE_K/2 = 16 uints.
// A fragment (ISA 7.12.2, 16-bit A 16x32): lane m = lane&15;
//   lanes 0-15:  V0-3 = K0..7,  V4-7 = K16..23
//   lanes 16-31: V0-3 = K8..15, V4-7 = K24..31
__device__ __forceinline__ v16bf load_frag_a(const unsigned int* blk, int lane) {
    const unsigned int* row = blk + (lane & 15) * (TILE_K / 2);
    const int hi4 = ((lane >> 4) & 1) * 4;
    union { v16bf v; unsigned int u[8]; } f;
#pragma unroll
    for (int r = 0; r < 4; ++r) f.u[r]     = row[hi4 + r];
#pragma unroll
    for (int r = 0; r < 4; ++r) f.u[4 + r] = row[8 + hi4 + r];
    return f.v;
}

// B fragment (same K order as A): column n = lane&15;
//   lanes 0-15:  V0-3 = K0..7,  V4-7 = K16..23
//   lanes 16-31: V0-3 = K8..15, V4-7 = K24..31
__device__ __forceinline__ v16bf load_frag_b(const unsigned int* blk, int lane) {
    const unsigned int* col = blk + (lane & 15) * (TILE_K / 2);
    const int kb = ((lane >> 4) & 1) * 4;
    union { v16bf v; unsigned int u[8]; } f;
#pragma unroll
    for (int r = 0; r < 4; ++r) f.u[r] = col[kb + r];
#pragma unroll
    for (int r = 0; r < 4; ++r) f.u[4 + r] = col[8 + kb + r];
    return f.v;
}

// C[GM x GN] = A[GM x GK] * B[GK x GN]; fp32 in/out, bf16 WMMA, fp32 accumulate.
// 256 threads = 8 wave32; C tile 128x128; wave owns a 32x64 sub-tile.
__global__ __launch_bounds__(256) void gemm_bf16_wmma(
    const float* __restrict__ A, const float* __restrict__ B, float* __restrict__ C) {
    __shared__ unsigned int As[TILE_M * TILE_K / 2];   // As[m][k/2]
    __shared__ unsigned int Bs[TILE_N * TILE_K / 2];   // Bs[n][k/2] (B transposed)

    const int t     = threadIdx.x;
    const int lane  = t & 31;
    const int wave  = t >> 5;                // 0..7
    const int m0    = blockIdx.y * TILE_M;
    const int n0    = blockIdx.x * TILE_N;
    const int m_off = (wave & 3) * 32;
    const int n_off = (wave >> 2) * 64;

    v8f acc[2][4];
#pragma unroll
    for (int i = 0; i < 2; ++i)
#pragma unroll
        for (int j = 0; j < 4; ++j)
            acc[i][j] = {};

    for (int k0 = 0; k0 < GK; k0 += TILE_K) {
        // A tile 128x32: pairs along k -> float2 load, one packed b32 LDS store
#pragma unroll
        for (int j = 0; j < 8; ++j) {
            int p = j * 256 + t;             // 0..2047 packed pairs
            int r = p >> 4;                  // m row
            int c = p & 15;                  // k pair index
            const float2 f = *(const float2*)(A + (size_t)(m0 + r) * GK + k0 + c * 2);
            As[r * (TILE_K / 2) + c] = pack_bf16(f.x, f.y);
        }
        // B tile 32x128 transposed: pair = rows k,k+1 at same n (coalesced across n)
#pragma unroll
        for (int j = 0; j < 8; ++j) {
            int p  = j * 256 + t;            // 0..2047
            int kp = p >> 7;                 // k pair index 0..15
            int c  = p & 127;                // n
            const float b0 = B[(size_t)(k0 + kp * 2) * GN + n0 + c];
            const float b1 = B[(size_t)(k0 + kp * 2 + 1) * GN + n0 + c];
            Bs[c * (TILE_K / 2) + kp] = pack_bf16(b0, b1);
        }
        __syncthreads();

        v16bf a0 = load_frag_a(&As[m_off * (TILE_K / 2)], lane);
        v16bf a1 = load_frag_a(&As[(m_off + 16) * (TILE_K / 2)], lane);
#pragma unroll
        for (int j = 0; j < 4; ++j) {
            v16bf b = load_frag_b(&Bs[(n_off + j * 16) * (TILE_K / 2)], lane);
            acc[0][j] = __builtin_amdgcn_wmma_f32_16x16x32_bf16(
                false, a0, false, b, (short)0, acc[0][j], false, false);
            acc[1][j] = __builtin_amdgcn_wmma_f32_16x16x32_bf16(
                false, a1, false, b, (short)0, acc[1][j], false, false);
        }
        __syncthreads();
    }

    // C/D layout: lanes 0-15: N=lane, M=vgpr; lanes 16-31: N=lane-16, M=vgpr+8.
    // GN is constant -> per-fragment base pointer + immediate offsets.
    const int ln = lane & 15;
    const int lh = (lane >> 4) & 1;
    float* Cbase = C + (size_t)(m0 + m_off + lh * 8) * GN + (n0 + n_off + ln);
#pragma unroll
    for (int i = 0; i < 2; ++i)
#pragma unroll
        for (int j = 0; j < 4; ++j) {
            float* Cp = Cbase + (size_t)(i * 16) * GN + j * 16;
#pragma unroll
            for (int v = 0; v < 8; ++v)
                Cp[(size_t)v * GN] = acc[i][j][v];
        }
}

// Per-(b,s) attention core. One block (256 threads) per token.
// comb may alias qp (q row fully read into LDS before comb row is written).
__global__ __launch_bounds__(256) void attn_core(
    const float* __restrict__ qp, const float* __restrict__ kp,
    const float* __restrict__ vp, const float* __restrict__ mask,
    float* __restrict__ comb, float* __restrict__ w_out) {
    __shared__ float q[DMODEL], k[DMODEL], v[DMODEL];
    __shared__ float w[HEADS * HEADS];
    __shared__ float ssum[2 * HEADS];

    const int bs = blockIdx.x;               // 0..B*S-1
    const int t  = threadIdx.x;              // 0..255
    const size_t row = (size_t)bs * DMODEL;

#pragma unroll
    for (int j = 0; j < DMODEL / 256; ++j) {
        int idx = j * 256 + t;
        q[idx] = qp[row + idx];
        k[idx] = kp[row + idx];
        v[idx] = vp[row + idx];
    }
    __syncthreads();

    // per-head sums for sum-normalization (32 threads: one head per tensor)
    if (t < 2 * HEADS) {
        const float* src = (t < HEADS) ? q : k;
        int h = t & (HEADS - 1);
        float s = 0.f;
        for (int d = 0; d < KSZ; ++d) s += src[h * KSZ + d];
        ssum[t] = s;
    }
    __syncthreads();

#pragma unroll
    for (int j = 0; j < DMODEL / 256; ++j) {
        int idx = j * 256 + t;
        int h = idx >> 7;                    // /KSZ
        q[idx] = q[idx] / (ssum[h] + EPS);
        k[idx] = k[idx] / (ssum[HEADS + h] + EPS);
    }
    __syncthreads();

    // logits[qi][ki] = qn[qi,:] . kn[ki,:]  (16x16, contraction over 128)
    const int qi = t >> 4, ki = t & 15;
    float acc = 0.f;
    for (int d = 0; d < KSZ; ++d) acc += q[qi * KSZ + d] * k[ki * KSZ + d];
    const float m = mask[(size_t)bs * (HEADS * HEADS) + t];
    w[t] = (acc + m * MASK_NEG) * 0.08838834764831845f;   // 1/sqrt(128)
    __syncthreads();

    // softmax over ki, one thread per row
    if (t < HEADS) {
        float mx = -INFINITY;
        for (int c = 0; c < HEADS; ++c) mx = fmaxf(mx, w[t * HEADS + c]);
        float s = 0.f;
        for (int c = 0; c < HEADS; ++c) {
            float e = expf(w[t * HEADS + c] - mx);
            w[t * HEADS + c] = e;
            s += e;
        }
        float inv = 1.f / s;
        for (int c = 0; c < HEADS; ++c) w[t * HEADS + c] *= inv;
    }
    __syncthreads();

    // second output: attn_weights [B,S,H,H]
    w_out[(size_t)bs * (HEADS * HEADS) + t] = w[t];

    // attn + combine heads: comb[bs, qi*128+d] = sum_ki w[qi,ki] * v[ki*128+d]
#pragma unroll
    for (int j = 0; j < DMODEL / 256; ++j) {
        int idx = j * 256 + t;
        int qh = idx >> 7, d = idx & (KSZ - 1);
        float o = 0.f;
#pragma unroll
        for (int kk = 0; kk < HEADS; ++kk)
            o += w[qh * HEADS + kk] * v[kk * KSZ + d];
        comb[row + idx] = o;
    }
}

extern "C" void kernel_launch(void* const* d_in, const int* in_sizes, int n_in,
                              void* d_out, int out_size, void* d_ws, size_t ws_size,
                              hipStream_t stream) {
    (void)in_sizes; (void)n_in; (void)out_size; (void)ws_size;
    const float* query = (const float*)d_in[0];
    const float* keyt  = (const float*)d_in[1];
    const float* value = (const float*)d_in[2];
    const float* mask  = (const float*)d_in[3];
    const float* Wq    = (const float*)d_in[4];
    const float* Wk    = (const float*)d_in[5];
    const float* Wv    = (const float*)d_in[6];
    const float* Wo    = (const float*)d_in[7];

    float* qproj = (float*)d_ws;                        // 128 MB, reused as `comb`
    float* kproj = qproj + (size_t)GM * GN;             // 128 MB
    float* vproj = kproj + (size_t)GM * GN;             // 128 MB
    float* out   = (float*)d_out;                       // [B,S,DMODEL]
    float* wout  = out + (size_t)GM * DMODEL;           // [B,S,H,H]

    dim3 grid(GN / TILE_N, GM / TILE_M);
    dim3 block(256);
    hipLaunchKernelGGL(gemm_bf16_wmma, grid, block, 0, stream, query, Wq, qproj);
    hipLaunchKernelGGL(gemm_bf16_wmma, grid, block, 0, stream, keyt,  Wk, kproj);
    hipLaunchKernelGGL(gemm_bf16_wmma, grid, block, 0, stream, value, Wv, vproj);

    hipLaunchKernelGGL(attn_core, dim3(GM), dim3(256), 0, stream,
                       qproj, kproj, vproj, mask, /*comb=*/qproj, wout);

    hipLaunchKernelGGL(gemm_bf16_wmma, grid, block, 0, stream, qproj, Wo, out);
}
